// SiameseModel_6837587936209
// MI455X (gfx1250) — matrix operand order repaired and checked by reading, not compiled, not run on hardware
//
#include <hip/hip_runtime.h>
#include <cstdint>
#include <cstddef>

// Problem sizes (match reference): VOCAB=50000, D_EMB=512, H=512, M=512, B=1024, L=64
#define B_     1024
#define L_     64
#define D_     512
#define H_     512
#define M_     512
#define BH_    (B_ * H_)          // 524288 elements per LSTM run state

typedef __attribute__((ext_vector_type(16))) __bf16 v16bf;
typedef __attribute__((ext_vector_type(8)))  __bf16 v8bf;
typedef __attribute__((ext_vector_type(8)))  float  v8f;

// gfx1250 async global->LDS path (ASYNCcnt), with graceful fallback
#if defined(__has_builtin)
#if __has_builtin(__builtin_amdgcn_global_load_async_to_lds_b128) && \
    __has_builtin(__builtin_amdgcn_s_wait_asynccnt)
#define USE_ASYNC_LDS 1
#endif
#endif
#ifndef USE_ASYNC_LDS
#define USE_ASYNC_LDS 0
#endif

#if USE_ASYNC_LDS
// Builtin signature (from hipcc diagnostic): param0 = int vector_size(16) * addrspace(1),
// param1 = same in addrspace(3), then two constant ints (offset, cpol).
typedef int gv4i __attribute__((vector_size(16)));
__device__ __forceinline__ void async_ld16(const unsigned short* g, unsigned short* l) {
    __builtin_amdgcn_global_load_async_to_lds_b128(
        (__attribute__((address_space(1))) gv4i*)(gv4i*)const_cast<unsigned short*>(g),
        (__attribute__((address_space(3))) gv4i*)(gv4i*)l,
        0, 0);
}
#endif

// float -> bf16 (round-to-nearest-even), kept as raw ushort to avoid scalar __bf16 codegen paths
__device__ __forceinline__ unsigned short f2bfu(float f) {
    unsigned u = __builtin_bit_cast(unsigned, f);
    return (unsigned short)((u + 0x7FFFu + ((u >> 16) & 1u)) >> 16);
}
__device__ __forceinline__ float sigm(float x) { return 1.0f / (1.0f + __expf(-x)); }

// ---------------------------------------------------------------- utility kernels
__global__ void k_zero_u32(uint32_t* __restrict__ p, int n) {
    int i = blockIdx.x * blockDim.x + threadIdx.x;
    if (i < n) p[i] = 0u;
}

// fp32 -> bf16 bulk convert (weights)
__global__ void k_f32_to_bf16(const float* __restrict__ src, unsigned short* __restrict__ dst, int n4) {
    int i = blockIdx.x * blockDim.x + threadIdx.x;
    if (i >= n4) return;
    const float4 v = *(const float4*)(src + (size_t)i * 4);
    *(ushort4*)(dst + (size_t)i * 4) =
        make_ushort4(f2bfu(v.x), f2bfu(v.y), f2bfu(v.z), f2bfu(v.w));
}

// embedding gather + fp32->bf16: out[b][t][d] = bf16(emb[tok[b*L+t]][d])
__global__ void k_embed_bf16(const int* __restrict__ tok, const float* __restrict__ emb,
                             unsigned short* __restrict__ out, int n4) {
    int i = blockIdx.x * blockDim.x + threadIdx.x;
    if (i >= n4) return;
    int e    = i * 4;
    int trow = e >> 9;        // (b,t) flat index (D=512)
    int d    = e & 511;
    int row  = tok[trow];
    const float* src = emb + (size_t)row * 512 + d;
    __builtin_prefetch(src + 128, 0, 0);   // gfx1250 global_prefetch_b8
    float4 v = *(const float4*)src;
    *(ushort4*)(out + e) = make_ushort4(f2bfu(v.x), f2bfu(v.y), f2bfu(v.z), f2bfu(v.w));
}

// ---------------------------------------------------------------- fused LSTM step
// One launch = one time step for all 4 runs (z: 0=fw/x1, 1=bw/x1, 2=fw/x2, 3=bw/x2).
// Block: 64 batch rows x 32 hidden cols. GEMM over K=1024 (x_t | h_prev) against the
// 4 gate column strips (i,j,f,o), WMMA bf16 16x16x32, fp32 accum, then the cell update.
// LDS staging is double-buffered and (when available) fed by GLOBAL_LOAD_ASYNC_TO_LDS_B128;
// the post-GEMM gate-exchange tile aliases the staging arena (separated by a barrier).
#define A_BYTES     (64 * 80 * 2)            // 10240
#define B_BYTES     (64 * 144 * 2)           // 18432
#define STAGE_BYTES (A_BYTES + B_BYTES)      // 28672 per buffer

__global__ __launch_bounds__(256) void k_lstm_step(
    const unsigned short* __restrict__ X1, const unsigned short* __restrict__ X2,
    const unsigned short* __restrict__ Wfw, const unsigned short* __restrict__ Wbw,
    const float* __restrict__ bias_fw, const float* __restrict__ bias_bw,
    const int* __restrict__ s1, const int* __restrict__ s2,
    float* __restrict__ C, float* __restrict__ Hf,
    const unsigned short* __restrict__ HbfIn, unsigned short* __restrict__ HbfOut,
    int t)
{
    __shared__ alignas(128) char smem[2 * STAGE_BYTES];   // 57344 B staging arena
    __shared__ int sTI[64];                               // per-row time index (reverse_sequence)
    __shared__ int sSL[64];                               // per-row seqlen

    const int tid = threadIdx.x;
    const int run = blockIdx.z;
    const int hc0 = blockIdx.x * 32;         // hidden col block (512/32 = 16)
    const int rb0 = blockIdx.y * 64;         // batch row block (1024/64 = 16)

    const unsigned short* Xp = (run < 2) ? X1 : X2;
    const unsigned short* Wp = (run & 1) ? Wbw : Wfw;
    const float* bias        = (run & 1) ? bias_bw : bias_fw;
    const int* slen          = (run < 2) ? s1 : s2;
    const int rev            = run & 1;

    float* Crun = C  + (size_t)run * BH_;
    float* Hrun = Hf + (size_t)run * BH_;
    const unsigned short* HinRun  = HbfIn  + (size_t)run * BH_;
    unsigned short*       HoutRun = HbfOut + (size_t)run * BH_;

    if (tid < 64) {
        int b  = rb0 + tid;
        int sl = slen[b];
        sSL[tid] = sl;
        sTI[tid] = rev ? ((t < sl) ? (sl - 1 - t) : t) : t;
    }
    __syncthreads();

    const int wave = tid >> 5;               // 8 waves (wave32)
    const int lane = tid & 31;
    const int wave_row = wave >> 2;          // 0..1 -> 32 rows each
    const int wave_col = wave & 3;           // 0..3 -> gate strip
    const int wr0  = wave_row * 32;
    const int wc0  = wave_col * 32;
    const int half = lane >> 4;
    const int l15  = lane & 15;

    // staging roles
    const int ar  = tid >> 2, aseg = tid & 3;   // A stage: row, 16-element K segment
    const int bkr = tid >> 2, bg   = tid & 3;   // B stage: K row, gate strip

    // hoisted per-thread source bases (time-index gather resolved once)
    const int ti = sTI[ar];
    const unsigned short* srcXrow = Xp + (((size_t)(rb0 + ar)) * L_ + ti) * 512 + aseg * 16;
    const unsigned short* srcHrow = HinRun + (size_t)(rb0 + ar) * 512 + aseg * 16;
    const unsigned short* srcB    = Wp + (size_t)bkr * 2048 + bg * 512 + hc0;

    auto stage = [&](int k0, int par) {
        unsigned short (*dA)[80]  = (unsigned short (*)[80]) (smem + par * STAGE_BYTES);
        unsigned short (*dB)[144] = (unsigned short (*)[144])(smem + par * STAGE_BYTES + A_BYTES);
        const unsigned short* asrc = (k0 < 512) ? (srcXrow + k0) : (srcHrow + (k0 - 512));
        const unsigned short* bsrc = srcB + (size_t)k0 * 2048;
#if USE_ASYNC_LDS
        async_ld16(asrc + 0,  &dA[ar][aseg * 16 + 0]);
        async_ld16(asrc + 8,  &dA[ar][aseg * 16 + 8]);
        async_ld16(bsrc + 0,  &dB[bkr][bg * 32 + 0]);
        async_ld16(bsrc + 8,  &dB[bkr][bg * 32 + 8]);
        async_ld16(bsrc + 16, &dB[bkr][bg * 32 + 16]);
        async_ld16(bsrc + 24, &dB[bkr][bg * 32 + 24]);
#else
        const uint4* a4 = (const uint4*)asrc;
        const uint4* b4 = (const uint4*)bsrc;
        uint4 a0 = a4[0], a1 = a4[1];
        uint4 w0 = b4[0], w1 = b4[1], w2 = b4[2], w3 = b4[3];
        *(uint4*)&dA[ar][aseg * 16 + 0] = a0;
        *(uint4*)&dA[ar][aseg * 16 + 8] = a1;
        uint4* d4 = (uint4*)&dB[bkr][bg * 32];
        d4[0] = w0; d4[1] = w1; d4[2] = w2; d4[3] = w3;
#endif
    };

    v8f acc[2][2];
    v8f zv = {0.f, 0.f, 0.f, 0.f, 0.f, 0.f, 0.f, 0.f};
    #pragma unroll
    for (int i = 0; i < 2; ++i)
        #pragma unroll
        for (int j = 0; j < 2; ++j) acc[i][j] = zv;

    // pipeline prologue: stage tile 0
    stage(0, 0);
#if USE_ASYNC_LDS
    __builtin_amdgcn_s_wait_asynccnt(0);
#endif
    __syncthreads();

    for (int kb = 0; kb < 16; ++kb) {
        const int cur = kb & 1;
        if (kb + 1 < 16) stage((kb + 1) * 64, (kb + 1) & 1);   // overlap next-tile fetch

        unsigned short (*sA)[80]  = (unsigned short (*)[80]) (smem + cur * STAGE_BYTES);
        unsigned short (*sB)[144] = (unsigned short (*)[144])(smem + cur * STAGE_BYTES + A_BYTES);

        #pragma unroll
        for (int ks = 0; ks < 64; ks += 32) {
            v16bf a[2], bb[2];
            // A fragment per ISA 16x32 layout: lane half h holds K in [h*8,h*8+8) U [h*8+16,h*8+24)
            #pragma unroll
            for (int rt = 0; rt < 2; ++rt) {
                int m  = wr0 + rt * 16 + l15;
                int kb_ = ks + half * 8;
                v8bf lo = *(const v8bf*)&sA[m][kb_];
                v8bf hi = *(const v8bf*)&sA[m][kb_ + 16];
                a[rt] = __builtin_shufflevector(lo, hi,
                        0,1,2,3,4,5,6,7,8,9,10,11,12,13,14,15);
            }
            // B fragment: lane column N=l15, lane half h holds K in [h*8,h*8+8) U [h*8+16,h*8+24)
            #pragma unroll
            for (int ct = 0; ct < 2; ++ct) {
                int col = wc0 + ct * 16 + l15;
                #pragma unroll
                for (int s = 0; s < 16; ++s) {
                    int kk = ks + half * 8 + (s < 8 ? s : s + 8);
                    bb[ct][s] = __builtin_bit_cast(__bf16, sB[kk][col]);
                }
            }
            #pragma unroll
            for (int rt = 0; rt < 2; ++rt)
                #pragma unroll
                for (int ct = 0; ct < 2; ++ct)
                    acc[rt][ct] = __builtin_amdgcn_wmma_f32_16x16x32_bf16(
                        false, a[rt], false, bb[ct], (short)0, acc[rt][ct], false, false);
        }

#if USE_ASYNC_LDS
        __builtin_amdgcn_s_wait_asynccnt(0);   // next tile resident before anyone reads it
#endif
        __syncthreads();                       // also: all waves done reading `cur` buffer
    }
    // loop-final barrier above separates last fragment reads from the aliased sC scatter

    // gate-exchange tile aliases the staging arena: [gate][row][hid] (33792 B <= 57344 B)
    float (*sC)[64][33] = (float (*)[64][33])smem;

    // scatter accumulators (C/D layout: M = half*8 + r, N = lane&15)
    #pragma unroll
    for (int rt = 0; rt < 2; ++rt)
        #pragma unroll
        for (int ct = 0; ct < 2; ++ct)
            #pragma unroll
            for (int r = 0; r < 8; ++r)
                sC[wave_col][wr0 + rt * 16 + half * 8 + r][ct * 16 + l15] = acc[rt][ct][r];
    __syncthreads();

    // fused LSTM cell update with TF dynamic_rnn masking (freeze state at t >= seqlen)
    #pragma unroll
    for (int q = 0; q < 8; ++q) {
        int idx = tid + q * 256;
        int row = idx >> 5;
        int hid = idx & 31;
        int b = rb0 + row;
        int n = hc0 + hid;
        float gi = sC[0][row][hid] + bias[n];
        float gj = sC[1][row][hid] + bias[512 + n];
        float gf = sC[2][row][hid] + bias[1024 + n];
        float go = sC[3][row][hid] + bias[1536 + n];
        size_t off = (size_t)b * 512 + n;
        float cold = Crun[off];
        float hold = Hrun[off];
        float cnew = cold * sigm(gf + 1.0f) + sigm(gi) * tanhf(gj);
        float hnew = tanhf(cnew) * sigm(go);
        bool act = (t < sSL[row]);
        float cw = act ? cnew : cold;
        float hw = act ? hnew : hold;
        Crun[off]    = cw;
        Hrun[off]    = hw;
        HoutRun[off] = f2bfu(hw);
    }
}

// ---------------------------------------------------------------- head
// h_dist[b] = sum((h1-h2)^2) where h1=[run0|run1], h2=[run2|run3]
__global__ __launch_bounds__(256) void k_hdist(const float* __restrict__ Hf, float* __restrict__ hd) {
    __shared__ float red[256];
    int b = blockIdx.x, tid = threadIdx.x;
    float s = 0.f;
    for (int k = tid; k < 1024; k += 256) {
        float a = (k < 512) ? Hf[(size_t)0 * BH_ + (size_t)b * 512 + k]
                            : Hf[(size_t)1 * BH_ + (size_t)b * 512 + (k - 512)];
        float c = (k < 512) ? Hf[(size_t)2 * BH_ + (size_t)b * 512 + k]
                            : Hf[(size_t)3 * BH_ + (size_t)b * 512 + (k - 512)];
        float d = a - c; s += d * d;
    }
    red[tid] = s; __syncthreads();
    for (int o = 128; o; o >>= 1) { if (tid < o) red[tid] += red[tid + o]; __syncthreads(); }
    if (tid == 0) hd[b] = red[0];
}

// e1 = relu([h1, h2, h_dist, h1*h2] @ W1 + b1); h_dist row handled as rank-1 add
__global__ __launch_bounds__(256) void k_mlp1(const float* __restrict__ Hf, const float* __restrict__ hd,
                                              const float* __restrict__ W1, const float* __restrict__ b1,
                                              float* __restrict__ e1) {
    int gid = blockIdx.x * 256 + threadIdx.x;   // B*M
    int b = gid >> 9, m = gid & 511;
    const float* h1f = Hf + (size_t)0 * BH_ + (size_t)b * 512;  // fw(x1)
    const float* h1b = Hf + (size_t)1 * BH_ + (size_t)b * 512;  // bw(x1)
    const float* h2f = Hf + (size_t)2 * BH_ + (size_t)b * 512;  // fw(x2)
    const float* h2b = Hf + (size_t)3 * BH_ + (size_t)b * 512;  // bw(x2)
    float acc = b1[m] + hd[b] * W1[(size_t)2048 * 512 + m];
    for (int k = 0; k < 512; ++k) {                             // h1[k], h2[k] (k<512)
        float h1 = h1f[k], h2 = h2f[k];
        acc += h1 * W1[(size_t)k * 512 + m];
        acc += h2 * W1[(size_t)(1024 + k) * 512 + m];
        acc += (h1 * h2) * W1[(size_t)(2049 + k) * 512 + m];
    }
    for (int k = 0; k < 512; ++k) {                             // h1[512+k], h2[512+k]
        float h1 = h1b[k], h2 = h2b[k];
        acc += h1 * W1[(size_t)(512 + k) * 512 + m];
        acc += h2 * W1[(size_t)(1536 + k) * 512 + m];
        acc += (h1 * h2) * W1[(size_t)(2561 + k) * 512 + m];
    }
    e1[gid] = fmaxf(acc, 0.0f);
}

__global__ __launch_bounds__(256) void k_mlp2(const float* __restrict__ e1, const float* __restrict__ W2,
                                              const float* __restrict__ b2, float* __restrict__ out) {
    int gid = blockIdx.x * 256 + threadIdx.x;   // B*2
    if (gid >= B_ * 2) return;
    int b = gid >> 1, u = gid & 1;
    float acc = b2[u];
    const float* e = e1 + (size_t)b * 512;
    for (int k = 0; k < 512; ++k) acc += e[k] * W2[k * 2 + u];
    out[gid] = acc;
}

// ---------------------------------------------------------------- launcher
extern "C" void kernel_launch(void* const* d_in, const int* in_sizes, int n_in,
                              void* d_out, int out_size, void* d_ws, size_t ws_size,
                              hipStream_t stream) {
    const int*   input1    = (const int*)d_in[0];
    const int*   input2    = (const int*)d_in[1];
    const int*   seqlen1   = (const int*)d_in[2];
    const int*   seqlen2   = (const int*)d_in[3];
    const float* embedding = (const float*)d_in[4];
    const float* k_fw      = (const float*)d_in[5];
    const float* b_fw      = (const float*)d_in[6];
    const float* k_bw      = (const float*)d_in[7];
    const float* b_bw      = (const float*)d_in[8];
    const float* W1        = (const float*)d_in[9];
    const float* b1        = (const float*)d_in[10];
    const float* W2        = (const float*)d_in[11];
    const float* b2        = (const float*)d_in[12];
    float* out = (float*)d_out;

    // workspace carve-up (~163 MB total)
    char* ws = (char*)d_ws;
    size_t off = 0;
    auto carve = [&](size_t bytes) -> char* {
        char* p = ws + off;
        off = (off + bytes + 255) & ~(size_t)255;
        return p;
    };
    unsigned short* X1  = (unsigned short*)carve((size_t)B_ * L_ * D_ * 2);   // 64 MB
    unsigned short* X2  = (unsigned short*)carve((size_t)B_ * L_ * D_ * 2);   // 64 MB
    unsigned short* Wfw = (unsigned short*)carve((size_t)1024 * 2048 * 2);    // 4 MB
    unsigned short* Wbw = (unsigned short*)carve((size_t)1024 * 2048 * 2);    // 4 MB
    float* Cst          = (float*)carve((size_t)4 * BH_ * 4);                 // 8 MB
    float* Hst          = (float*)carve((size_t)4 * BH_ * 4);                 // 8 MB
    unsigned short* Hbf = (unsigned short*)carve((size_t)2 * 4 * BH_ * 2);    // 8 MB (double-buffered)
    float* hd           = (float*)carve((size_t)B_ * 4);
    float* e1           = (float*)carve((size_t)B_ * M_ * 4);                 // 2 MB
    const size_t HBF_HALF = (size_t)4 * BH_;   // bf16 elements per parity buffer

    // zero recurrent state (c, h, both h-bf16 parity buffers)
    {
        int nC = 4 * BH_;                       // fp32 words
        k_zero_u32<<<(nC + 255) / 256, 256, 0, stream>>>((uint32_t*)Cst, nC);
        k_zero_u32<<<(nC + 255) / 256, 256, 0, stream>>>((uint32_t*)Hst, nC);
        int nH = 4 * BH_;                       // 8*BH_ ushorts = 4*BH_ words
        k_zero_u32<<<(nH + 255) / 256, 256, 0, stream>>>((uint32_t*)Hbf, nH);
    }

    // weights fp32 -> bf16
    {
        int n4 = 1024 * 2048 / 4;
        k_f32_to_bf16<<<(n4 + 255) / 256, 256, 0, stream>>>(k_fw, Wfw, n4);
        k_f32_to_bf16<<<(n4 + 255) / 256, 256, 0, stream>>>(k_bw, Wbw, n4);
    }

    // embedding gather -> bf16 activations
    {
        int n4 = B_ * L_ * D_ / 4;
        k_embed_bf16<<<(n4 + 255) / 256, 256, 0, stream>>>(input1, embedding, X1, n4);
        k_embed_bf16<<<(n4 + 255) / 256, 256, 0, stream>>>(input2, embedding, X2, n4);
    }

    // 64 recurrent steps; all 4 LSTM runs per launch; h double-buffered across launches
    for (int t = 0; t < L_; ++t) {
        const unsigned short* hin = Hbf + (size_t)(t & 1) * HBF_HALF;
        unsigned short* hout      = Hbf + (size_t)((t + 1) & 1) * HBF_HALF;
        k_lstm_step<<<dim3(16, 16, 4), 256, 0, stream>>>(
            X1, X2, Wfw, Wbw, b_fw, b_bw, seqlen1, seqlen2,
            Cst, Hst, hin, hout, t);
    }

    // head
    k_hdist<<<B_, 256, 0, stream>>>(Hst, hd);
    k_mlp1<<<(B_ * M_) / 256, 256, 0, stream>>>(Hst, hd, W1, b1, e1);
    k_mlp2<<<(B_ * 2 + 255) / 256, 256, 0, stream>>>(e1, W2, b2, out);
}
